// SelfAttention_37769942401291
// MI455X (gfx1250) — matrix operand order repaired and checked by reading, not compiled, not run on hardware
//
#include <hip/hip_runtime.h>
#include <math.h>

#define B_      8
#define C_      8
#define T_      256
#define DIN     1536
#define DOUT    512
#define MTOT    (B_ * C_ * T_)   /* 16384 */
#define TFLAT   (C_ * T_)        /* 2048  */
#define RADIUS  13
#define NEG_INF (-3.4028234663852886e38f)

typedef __bf16 bf16_t;
typedef bf16_t v8bf  __attribute__((ext_vector_type(8)));
typedef bf16_t v16bf __attribute__((ext_vector_type(16)));
typedef float  v8f   __attribute__((ext_vector_type(8)));
typedef unsigned int u32x4 __attribute__((ext_vector_type(4)));
typedef int          i32x4 __attribute__((ext_vector_type(4)));
typedef int          i32x8 __attribute__((ext_vector_type(8)));

__device__ __forceinline__ bf16_t f2bf(float x) {
  union { float f; unsigned int u; } v; v.f = x;
  unsigned int r = v.u + 0x7fffu + ((v.u >> 16) & 1u);   // RNE
  unsigned short h = (unsigned short)(r >> 16);
  bf16_t out;
  __builtin_memcpy(&out, &h, sizeof(out));
  return out;
}

__device__ __forceinline__ unsigned pack_bf16x2(float a, float b) {
  union { float f; unsigned u; } x, y; x.f = a; y.f = b;
  unsigned lo = (x.u + 0x7fffu + ((x.u >> 16) & 1u)) >> 16;
  unsigned hi = (y.u + 0x7fffu + ((y.u >> 16) & 1u)) >> 16;
  return lo | (hi << 16);
}

// ---------------------------------------------------------------------------
// Kernel 0: one-time W1 [K=1536][N=512] f32 -> W1T [N=512][K=1536] bf16.
// LDS-tiled 32x32 transpose, coalesced both sides.
// ---------------------------------------------------------------------------
__global__ __launch_bounds__(256)
void w1_transpose_kernel(const float* __restrict__ W1, bf16_t* __restrict__ W1T) {
  __shared__ float tile[32][33];
  const int k0 = blockIdx.x * 32;          // 48 blocks
  const int n0 = blockIdx.y * 32;          // 16 blocks
  const int tx = threadIdx.x & 31, ty = threadIdx.x >> 5;
  for (int r = ty; r < 32; r += 8)
    tile[r][tx] = W1[(size_t)(k0 + r) * DOUT + n0 + tx];
  __syncthreads();
  for (int r = ty; r < 32; r += 8)
    W1T[(size_t)(n0 + r) * DIN + k0 + tx] = f2bf(tile[tx][r]);
}

// ---------------------------------------------------------------------------
// TDM: load one 512(N) x 32(K) bf16 tile of W1T into LDS.
// Pad descriptor: 64B row (tile_dim0=32 elems) + 16B pad -> LDS row stride 80B
// (== KPAD=40 elements), i.e. hardware reproduces the bank-spread layout.
// ---------------------------------------------------------------------------
#define KC   32
#define KPAD 40

__device__ __forceinline__ void tdm_load_w1t(const bf16_t* gsrc, unsigned lds_addr) {
  unsigned long long ga = (unsigned long long)(size_t)gsrc;
  u32x4 g0;
  g0[0] = 1u;                                        // count=1 (valid), user mode
  g0[1] = lds_addr;                                  // LDS byte address
  g0[2] = (unsigned)(ga & 0xffffffffu);              // global_addr[31:0]
  g0[3] = (unsigned)((ga >> 32) & 0x1ffffffu) | (2u << 30);  // addr[56:32] | type=2
  i32x8 g1;
  g1[0] = (1 << 16)        /* data_size = 2B            */
        | (1 << 20)        /* pad_enable                */
        | (3 << 22)        /* pad_interval: 16 DW = 64B */
        | (3 << 25);       /* pad_amount:    4 DW = 16B */
  g1[1] = (DIN & 0xffff) << 16;                      // tensor_dim0[15:0]
  g1[2] = (DIN >> 16) | ((DOUT & 0xffff) << 16);     // tensor_dim0[31:16] | dim1[15:0]
  g1[3] = (DOUT >> 16) | (KC << 16);                 // dim1[31:16] | tile_dim0=32
  g1[4] = DOUT;                                      // tile_dim1=512, tile_dim2=0
  g1[5] = DIN;                                       // tensor_dim0_stride[31:0]
  g1[6] = 0;                                         // stride0[47:32] | stride1[15:0]
  g1[7] = 0;                                         // stride1[47:16]
  i32x4 z4; z4[0] = 0; z4[1] = 0; z4[2] = 0; z4[3] = 0;
#if __clang_major__ >= 23
  i32x8 z8; z8[0]=0; z8[1]=0; z8[2]=0; z8[3]=0; z8[4]=0; z8[5]=0; z8[6]=0; z8[7]=0;
  __builtin_amdgcn_tensor_load_to_lds(g0, g1, z4, z4, z8, 0);
#else
  __builtin_amdgcn_tensor_load_to_lds(g0, g1, z4, z4, 0);
#endif
}

// ---------------------------------------------------------------------------
// Kernel 1: raw[m] = V(tanh(q[m,:] @ W1 + b1)) + Vb.
// Block = 256 thr (8 waves) owns 16 rows; wave owns 4 N-tiles (64 cols).
// Per 32-K chunk: TDM (double-buffered, tensorcnt-pipelined) fetches the W1T
// tile; lanes stage 16x32 query as bf16; 4x v_wmma_f32_16x16x32_bf16.
// ---------------------------------------------------------------------------
#define LDS_QS_ELEMS (16 * KPAD)
#define LDS_WT_ELEMS (DOUT * KPAD)
#define GEMM_LDS_BYTES ((LDS_QS_ELEMS + 2 * LDS_WT_ELEMS) * 2 + 64)

__global__ __launch_bounds__(256)
void score_gemm_kernel(const float* __restrict__ q,
                       const bf16_t* __restrict__ W1T,
                       const float* __restrict__ b1,
                       const float* __restrict__ Vw,
                       const float* __restrict__ Vb,
                       float* __restrict__ raw) {
  extern __shared__ char smem[];
  bf16_t* qs  = (bf16_t*)smem;                       // 16 x KPAD
  bf16_t* wt0 = qs + LDS_QS_ELEMS;                   // 512 x KPAD (buf 0)
  bf16_t* wt1 = wt0 + LDS_WT_ELEMS;                  // 512 x KPAD (buf 1)
  float*  raw_s = (float*)(wt1 + LDS_WT_ELEMS);

  const int tid  = threadIdx.x;
  const int lane = tid & 31;
  const int wave = tid >> 5;
  const int half = lane >> 4;
  const int l16  = lane & 15;
  const int m0   = blockIdx.x * 16;

  const unsigned wt0_a = (unsigned)(size_t)(void*)wt0;
  const unsigned wt1_a = (unsigned)(size_t)(void*)wt1;

  if (tid < 16) raw_s[tid] = 0.f;

  // prologue: DMA chunk 0 into wt0
  if (wave == 0) tdm_load_w1t(W1T, wt0_a);

  v8f acc[4] = {v8f{}, v8f{}, v8f{}, v8f{}};

  const int NCH = DIN / KC;                          // 48
  for (int i = 0; i < NCH; ++i) {
    bf16_t* wtb = (i & 1) ? wt1 : wt0;
    __syncthreads();                                 // compute(i-1) done -> buffers free
    if (wave == 0 && i + 1 < NCH)
      tdm_load_w1t(W1T + (size_t)(i + 1) * KC, ((i + 1) & 1) ? wt1_a : wt0_a);
    // stage query chunk i (f32 -> bf16), vectorized: 128 lanes x float4
    if (tid < 128) {
      int r = tid >> 3, kv = tid & 7;
      float4 v = *(const float4*)(q + (size_t)(m0 + r) * DIN + i * KC + kv * 4);
      uint2 p; p.x = pack_bf16x2(v.x, v.y); p.y = pack_bf16x2(v.z, v.w);
      *(uint2*)(qs + r * KPAD + kv * 4) = p;
    }
    if (wave == 0) {
      if (i + 1 < NCH) __builtin_amdgcn_s_wait_tensorcnt(1);  // chunk i done, i+1 may fly
      else             __builtin_amdgcn_s_wait_tensorcnt(0);  // drain
    }
    __syncthreads();                                 // qs stores + TDM(i) visible

    // A fragment (ISA 16-bit A 16x32):
    // lanes 0-15: K 0..7,16..23 ; lanes 16-31: K 8..15,24..31
    v8bf alo = *(const v8bf*)(qs + l16 * KPAD + half * 8);
    v8bf ahi = *(const v8bf*)(qs + l16 * KPAD + 16 + half * 8);
    v16bf a = __builtin_shufflevector(alo, ahi,
              0,1,2,3,4,5,6,7,8,9,10,11,12,13,14,15);
#pragma unroll
    for (int t = 0; t < 4; ++t) {
      const bf16_t* wp = wtb + (size_t)(wave * 64 + t * 16 + l16) * KPAD;
      v8bf blo = *(const v8bf*)(wp + half * 8);
      v8bf bhi = *(const v8bf*)(wp + 16 + half * 8);
      v16bf b = __builtin_shufflevector(blo, bhi,
                0,1,2,3,4,5,6,7,8,9,10,11,12,13,14,15);
      acc[t] = __builtin_amdgcn_wmma_f32_16x16x32_bf16(
                 false, a, false, b, (short)0, acc[t], false, false);
    }
  }

  // Epilogue: tanh(acc + b1[n]) * Vw[n], reduce over N.
  float partial[8];
#pragma unroll
  for (int j = 0; j < 8; ++j) partial[j] = 0.f;
#pragma unroll
  for (int t = 0; t < 4; ++t) {
    int n = wave * 64 + t * 16 + l16;
    float bb = b1[n], vw = Vw[n];
#pragma unroll
    for (int j = 0; j < 8; ++j)
      partial[j] += tanhf(acc[t][j] + bb) * vw;
  }
#pragma unroll
  for (int off = 1; off < 16; off <<= 1) {
#pragma unroll
    for (int j = 0; j < 8; ++j)
      partial[j] += __shfl_xor(partial[j], off, 32);
  }
  if (l16 == 0) {
#pragma unroll
    for (int j = 0; j < 8; ++j)
      atomicAdd(&raw_s[half * 8 + j], partial[j]);
  }
  __syncthreads();
  if (tid < 16) raw[m0 + tid] = raw_s[tid] + Vb[0];
}

// ---------------------------------------------------------------------------
// Kernel 2: greedy temporal NMS per cine (stable rank == jnp.argsort(-s)),
// then sigmoid. One block per cine.
// ---------------------------------------------------------------------------
__global__ __launch_bounds__(256)
void suppress_kernel(const float* __restrict__ raw, float* __restrict__ probs) {
  __shared__ float sv[T_];
  __shared__ short order[T_];
  __shared__ unsigned char mk[T_];
  const int tid = threadIdx.x;
  const float s = raw[(size_t)blockIdx.x * T_ + tid];
  sv[tid] = s; mk[tid] = 1;
  __syncthreads();
  int rank = 0;
  for (int j = 0; j < T_; ++j) {
    float o = sv[j];
    rank += (o > s) || (o == s && j < tid);
  }
  order[rank] = (short)tid;
  __syncthreads();
  for (int t = 0; t < T_; ++t) {
    int i = order[t];
    bool keep = mk[i] != 0;            // mk[i] never written this round
    if (keep) {
      int d = tid - i;
      if (d != 0 && d >= -RADIUS && d <= RADIUS) mk[tid] = 0;
    }
    if (tid == i) {
      float val = keep ? sv[i] : NEG_INF;
      probs[(size_t)blockIdx.x * T_ + i] = 1.0f / (1.0f + __expf(-val));
    }
    __syncthreads();
  }
}

// ---------------------------------------------------------------------------
// Kernel 3: per-batch top-p(0.7) mask (stable descending bitonic sort),
// normalize, write attn + masked_score, zero context. One block per batch.
// ---------------------------------------------------------------------------
__global__ __launch_bounds__(256)
void topp_kernel(const float* __restrict__ probs,
                 float* __restrict__ attn,
                 float* __restrict__ masked,
                 float* __restrict__ context) {
  __shared__ float key[TFLAT];
  __shared__ int   idx[TFLAT];
  __shared__ float mval[TFLAT];
  __shared__ float sA[256], sB[256];
  __shared__ float s_ssum;
  const int tid = threadIdx.x, b = blockIdx.x;
  const float* pb = probs + (size_t)b * TFLAT;

  for (int i = tid; i < TFLAT; i += 256) {
    float p = pb[i];
    p = isfinite(p) ? fmaxf(p, 0.f) : 0.f;
    key[i] = p; idx[i] = i;
  }
  for (int i = tid; i < DIN; i += 256) context[(size_t)b * DIN + i] = 0.f;
  if (tid == 0) s_ssum = 0.f;
  __syncthreads();

  for (int kk = 2; kk <= TFLAT; kk <<= 1) {
    for (int jj = kk >> 1; jj > 0; jj >>= 1) {
      for (int i = tid; i < TFLAT; i += 256) {
        int ixj = i ^ jj;
        if (ixj > i) {
          bool up = ((i & kk) == 0);
          float pi = key[i], pj = key[ixj];
          int ii = idx[i], ij = idx[ixj];
          bool bef = (pj > pi) || (pj == pi && ij < ii);
          if (bef == up) { key[i] = pj; key[ixj] = pi; idx[i] = ij; idx[ixj] = ii; }
        }
      }
      __syncthreads();
    }
  }

  float loc[8]; float run = 0.f;
  const int base = tid * 8;
#pragma unroll
  for (int m = 0; m < 8; ++m) { run += key[base + m]; loc[m] = run; }
  sA[tid] = run;
  __syncthreads();
  float* src = sA; float* dst = sB;
  for (int off = 1; off < 256; off <<= 1) {
    float v = src[tid];
    if (tid >= off) v += src[tid - off];
    dst[tid] = v;
    __syncthreads();
    float* tmp = src; src = dst; dst = tmp;
  }
  float inc = src[tid];
  float total = src[255];
  float excl = inc - run;
  __syncthreads();

  float thr = 0.7f * (total + 1e-8f);
  float ssum_part = 0.f;
#pragma unroll
  for (int m = 0; m < 8; ++m) {
    int r = base + m;
    bool keep = (excl + loc[m] <= thr) || (r < 3);
    float mv = keep ? key[r] : 0.f;
    mval[idx[r]] = mv;
    ssum_part += mv;
  }
  atomicAdd(&s_ssum, ssum_part);
  __syncthreads();

  float ssum = s_ssum;
  float invs = 1.0f / (ssum + 1e-8f);
  bool fb = (ssum <= 0.f);
  for (int i = tid; i < TFLAT; i += 256) {
    float mv = mval[i];
    masked[(size_t)b * TFLAT + i] = mv;
    attn[(size_t)b * TFLAT + i] = fb ? (1.0f / TFLAT) : mv * invs;
  }
}

// ---------------------------------------------------------------------------
// Kernel 4: context[b,d] = sum_ct attn[b,ct] * values[b,ct,d]. Bandwidth-
// bound; split over 8 CT-chunks with global f32 atomics (context zeroed by K3).
// ---------------------------------------------------------------------------
__global__ __launch_bounds__(256)
void context_kernel(const float* __restrict__ values,
                    const float* __restrict__ attn,
                    float* __restrict__ context) {
  __shared__ float att[256];
  const int tid = threadIdx.x;
  const int dc = blockIdx.x;   // 0..5
  const int cc = blockIdx.y;   // 0..7
  const int b  = blockIdx.z;   // 0..7
  att[tid] = attn[(size_t)b * TFLAT + cc * 256 + tid];
  __syncthreads();
  const int d = dc * 256 + tid;
  const float* vp = values + ((size_t)b * TFLAT + (size_t)cc * 256) * DIN + d;
  float acc = 0.f;
#pragma unroll 8
  for (int j = 0; j < 256; ++j) acc += att[j] * vp[(size_t)j * DIN];
  atomicAdd(&context[(size_t)b * DIN + d], acc);
}

// ---------------------------------------------------------------------------
extern "C" void kernel_launch(void* const* d_in, const int* in_sizes, int n_in,
                              void* d_out, int out_size, void* d_ws, size_t ws_size,
                              hipStream_t stream) {
  (void)in_sizes; (void)n_in; (void)out_size; (void)ws_size;
  const float* q   = (const float*)d_in[0];
  const float* val = (const float*)d_in[1];
  const float* W1  = (const float*)d_in[2];
  const float* b1  = (const float*)d_in[3];
  const float* Vw  = (const float*)d_in[4];
  const float* Vb  = (const float*)d_in[5];

  float* out     = (float*)d_out;
  float* context = out;                                   // [B, DIN]
  float* attn    = out + (size_t)B_ * DIN;                // [B, C, T, 1]
  float* masked  = attn + (size_t)B_ * TFLAT;             // [B, C, T, 1]

  float*  raw   = (float*)d_ws;                           // [MTOT] f32
  float*  probs = raw + MTOT;                             // [MTOT] f32
  bf16_t* W1T   = (bf16_t*)((char*)d_ws + 2 * MTOT * sizeof(float)); // [DOUT][DIN] bf16

  w1_transpose_kernel<<<dim3(DIN / 32, DOUT / 32), 256, 0, stream>>>(W1, W1T);
  score_gemm_kernel<<<MTOT / 16, 256, GEMM_LDS_BYTES, stream>>>(q, W1T, b1, Vw, Vb, raw);
  suppress_kernel<<<B_ * C_, 256, 0, stream>>>(raw, probs);
  topp_kernel<<<B_, 256, 0, stream>>>(probs, attn, masked, context);
  context_kernel<<<dim3(DIN / 256, TFLAT / 256, B_), 256, 0, stream>>>(val, attn, context);
}
